// HGT_39281770889881
// MI455X (gfx1250) — matrix operand order repaired and checked by reading, not compiled, not run on hardware
//
#include <hip/hip_runtime.h>
#include <math.h>

// ---------------- problem constants (match reference) ----------------
#define NI_C 32768LL
#define NV_C 131072LL
#define EVV_C 1048576LL
#define EVI_C 524288LL
#define EIV_C 524288LL
#define L_C 6
constexpr float INV_SQRT_D_C = 0.35355339059327373f; // 1/sqrt(8)

typedef __attribute__((ext_vector_type(16))) _Float16 v16h;
typedef __attribute__((ext_vector_type(8)))  float    v8f;
typedef unsigned int u32x4 __attribute__((ext_vector_type(4)));
typedef int          i32x4 __attribute__((ext_vector_type(4)));
typedef int          i32x8 __attribute__((ext_vector_type(8)));

// ---------------- device helpers ----------------
__device__ __forceinline__ float gelu_exact(float x) {
  return 0.5f * x * (1.0f + erff(x * 0.70710678118654752f));
}
__device__ __forceinline__ unsigned fkey(float f) {
  unsigned u = __float_as_uint(f);
  return (u & 0x80000000u) ? ~u : (u | 0x80000000u);
}
__device__ __forceinline__ float funkey(unsigned k) {
  unsigned u = (k & 0x80000000u) ? (k & 0x7fffffffu) : ~k;
  return __uint_as_float(u);
}

// ---------------- generic fill ----------------
__global__ void fill_f32(float* __restrict__ p, long long n, float v) {
  long long i = (long long)blockIdx.x * blockDim.x + threadIdx.x;
  long long stride = (long long)gridDim.x * blockDim.x;
  for (; i < n; i += stride) p[i] = v;
}

// ---------------- WMMA GEMM: Y[N,Cout] = op(X[N,64]) @ W[Cout,64]^T + bias --------
// Each block: 64 rows x 64 cols; each wave: one 16-col strip, 4 row tiles.
// B operand (32x16 f16, ISA 7.12.2): lane n=lane&15 holds col (col0+n); halves 0..7 = K[kb+8*hi..+7], 8..15 = K[kb+16+8*hi..+7]
// A operand (16x32 f16): lane m=lane&15 holds row; halves 0..7 = K[kb+8*hi..+7],
//                        halves 8..15 = K[kb+16+8*hi..+7]
// C/D (16x16 f32): element r -> row rbase + r + 8*hi, col col0 + (lane&15)
template<bool GELU_IN, bool SKIP>
__global__ __launch_bounds__(128)
void gemm64_wmma(const float* __restrict__ X, const float* __restrict__ W,
                 const float* __restrict__ bias, float* __restrict__ Y,
                 const float* __restrict__ Xold, const float* __restrict__ skipp,
                 float* __restrict__ Y2, int ystride)
{
  const int lane = threadIdx.x & 31;
  const int wave = threadIdx.x >> 5;
  const int m    = lane & 15;
  const int hi   = lane >> 4;
  const long long row0 = (long long)blockIdx.x * 64;
  const int col0 = (blockIdx.y * 4 + wave) * 16;

  // ---- load the weight fragment once (K = 64 split into two 32-chunks) ----
  v16h B0, B1;
  {
    const float* br = W + (long long)(col0 + m) * 64 + hi * 8;
#pragma unroll
    for (int t = 0; t < 16; ++t) B0[t] = (_Float16)br[t + (t >= 8 ? 8 : 0)];
#pragma unroll
    for (int t = 0; t < 16; ++t) B1[t] = (_Float16)br[32 + t + (t >= 8 ? 8 : 0)];
  }
  const int colo = col0 + m;
  const float bcol = bias[colo];
  float blend = 0.0f;
  if (SKIP) blend = 1.0f / (1.0f + expf(-skipp[0]));

#pragma unroll
  for (int rt = 0; rt < 4; ++rt) {
    const long long rbase = row0 + rt * 16;
    v16h A0, A1;
    {
      const float* ar = X + (rbase + m) * 64 + hi * 8;
#pragma unroll
      for (int j = 0; j < 8; ++j) {
        float a0 = ar[j],      a1 = ar[16 + j];
        float a2 = ar[32 + j], a3 = ar[48 + j];
        if (GELU_IN) {
          a0 = gelu_exact(a0); a1 = gelu_exact(a1);
          a2 = gelu_exact(a2); a3 = gelu_exact(a3);
        }
        A0[j] = (_Float16)a0; A0[8 + j] = (_Float16)a1;
        A1[j] = (_Float16)a2; A1[8 + j] = (_Float16)a3;
      }
    }
    v8f acc = {};
    acc = __builtin_amdgcn_wmma_f32_16x16x32_f16(false, A0, false, B0, (short)0, acc, false, false);
    acc = __builtin_amdgcn_wmma_f32_16x16x32_f16(false, A1, false, B1, (short)0, acc, false, false);

#pragma unroll
    for (int r = 0; r < 8; ++r) {
      long long row = rbase + r + 8 * hi;
      float val = acc[r] + bcol;
      if (SKIP) val = blend * val + (1.0f - blend) * Xold[row * 64 + colo];
      Y[row * (long long)ystride + colo] = val;
      if (SKIP) { if (Y2) Y2[row * (long long)ystride + colo] = val; }
    }
  }
}

// ---------------- relation transform: out[n,h,:] = in[n,h,:] @ rel[h] (8x8) -------
__global__ __launch_bounds__(256)
void rel_transform(const float* __restrict__ k, const float* __restrict__ v,
                   const float* __restrict__ arel, const float* __restrict__ mrel,
                   float* __restrict__ krel, float* __restrict__ vrel, long long N)
{
  __shared__ float sA[512];
  __shared__ float sM[512];
  for (int i = threadIdx.x; i < 512; i += blockDim.x) { sA[i] = arel[i]; sM[i] = mrel[i]; }
  __syncthreads();
  long long t = (long long)blockIdx.x * blockDim.x + threadIdx.x;
  if (t >= N * 8) return;
  long long n = t >> 3;
  int h = (int)(t & 7);
  const float* kr = k + n * 64 + h * 8;
  const float* vr = v + n * 64 + h * 8;
  float ko[8] = {}, vo[8] = {};
#pragma unroll
  for (int d = 0; d < 8; ++d) {
    float kd = kr[d], vd = vr[d];
    const float* ra = sA + h * 64 + d * 8;
    const float* rm = sM + h * 64 + d * 8;
#pragma unroll
    for (int e = 0; e < 8; ++e) { ko[e] += kd * ra[e]; vo[e] += vd * rm[e]; }
  }
  float* kw = krel + n * 64 + h * 8;
  float* vw = vrel + n * 64 + h * 8;
#pragma unroll
  for (int e = 0; e < 8; ++e) { kw[e] = ko[e]; vw[e] = vo[e]; }
}

// ---------------- edge pass 1: logits + per-(dst,head) atomic max ----------------
__global__ __launch_bounds__(256)
void edge_alpha(const long long* __restrict__ src, const long long* __restrict__ dst,
                const float* __restrict__ q, const float* __restrict__ krel,
                const float* __restrict__ prel, float* __restrict__ alpha,
                unsigned* __restrict__ mkey, long long E)
{
  long long t = (long long)blockIdx.x * blockDim.x + threadIdx.x;
  if (t >= E * 8) return;
  long long e = t >> 3;
  int h = (int)(t & 7);
  if (h == 0) {  // prefetch the streamed index arrays ahead (global_prefetch_b8)
    __builtin_prefetch(src + e + 4096, 0, 0);
    __builtin_prefetch(dst + e + 4096, 0, 0);
  }
  long long s = src[e], d = dst[e];
  const float4* qd = (const float4*)(q + d * 64 + h * 8);
  const float4* ks = (const float4*)(krel + s * 64 + h * 8);
  float4 q0 = qd[0], q1 = qd[1];
  float4 k0 = ks[0], k1 = ks[1];
  float a = q0.x * k0.x + q0.y * k0.y + q0.z * k0.z + q0.w * k0.w
          + q1.x * k1.x + q1.y * k1.y + q1.z * k1.z + q1.w * k1.w;
  a *= prel[h] * INV_SQRT_D_C;
  alpha[e * 8 + h] = a;
  atomicMax(mkey + d * 8 + h, fkey(a));
}

// ---------------- decode max keys, zero the exp-sum buffer ----------------
__global__ void finalize_max(unsigned* __restrict__ mkey, float* __restrict__ sbuf, long long n) {
  long long i = (long long)blockIdx.x * blockDim.x + threadIdx.x;
  if (i >= n) return;
  unsigned k = mkey[i];
  float m = (k == 0u) ? 0.0f : funkey(k);     // key==0 -> no edges -> ref maps -inf to 0
  if (!isfinite(m)) m = 0.0f;
  ((float*)mkey)[i] = m;                       // reuse buffer as float m
  sbuf[i] = 0.0f;
}

// ---------------- edge pass 2: exp(logit - m), atomic sum per (dst,head) ---------
__global__ __launch_bounds__(256)
void edge_expsum(const long long* __restrict__ dst, float* __restrict__ alpha,
                 const float* __restrict__ mbuf, float* __restrict__ sbuf, long long E)
{
  long long t = (long long)blockIdx.x * blockDim.x + threadIdx.x;
  if (t >= E * 8) return;
  long long e = t >> 3;
  int h = (int)(t & 7);
  if (h == 0) __builtin_prefetch(dst + e + 4096, 0, 0);
  long long d = dst[e];
  float ew = expf(alpha[e * 8 + h] - mbuf[d * 8 + h]);
  alpha[e * 8 + h] = ew;
  atomicAdd(sbuf + d * 8 + h, ew);
}

// ---------------- edge pass 3: weighted scatter of v_rel into agg ----------------
__global__ __launch_bounds__(256)
void edge_agg(const long long* __restrict__ src, const long long* __restrict__ dst,
              const float* __restrict__ alpha, const float* __restrict__ sbuf,
              const float* __restrict__ vrel, float* __restrict__ agg, long long E)
{
  long long t = (long long)blockIdx.x * blockDim.x + threadIdx.x;
  if (t >= E * 8) return;
  long long e = t >> 3;
  int h = (int)(t & 7);
  if (h == 0) {
    __builtin_prefetch(src + e + 4096, 0, 0);
    __builtin_prefetch(dst + e + 4096, 0, 0);
  }
  long long s = src[e], d = dst[e];
  float w = alpha[e * 8 + h] / (sbuf[d * 8 + h] + 1e-16f);
  const float4* vs = (const float4*)(vrel + s * 64 + h * 8);
  float4 v0 = vs[0], v1 = vs[1];
  float* ad = agg + d * 64 + h * 8;
  atomicAdd(ad + 0, v0.x * w); atomicAdd(ad + 1, v0.y * w);
  atomicAdd(ad + 2, v0.z * w); atomicAdd(ad + 3, v0.w * w);
  atomicAdd(ad + 4, v1.x * w); atomicAdd(ad + 5, v1.y * w);
  atomicAdd(ad + 6, v1.z * w); atomicAdd(ad + 7, v1.w * w);
}

// ---------------- jumping-knowledge attention over S=5 stacked layers -----------
__global__ __launch_bounds__(128)
void jk_attention(const float* __restrict__ qkv, float* __restrict__ obuf)
{
  long long t = (long long)blockIdx.x * blockDim.x + threadIdx.x;
  if (t >= NI_C * 8) return;
  long long n = t >> 3;
  int h = (int)(t & 7);
  float q[5][8], k[5][8], v[5][8];
#pragma unroll
  for (int s = 0; s < 5; ++s) {
    const float* base = qkv + ((long long)s * NI_C + n) * 192 + h * 8;
#pragma unroll
    for (int j = 0; j < 8; ++j) { q[s][j] = base[j]; k[s][j] = base[64 + j]; v[s][j] = base[128 + j]; }
  }
#pragma unroll
  for (int s = 0; s < 5; ++s) {
    float sc[5];
    float mx = -1e30f;
#pragma unroll
    for (int u = 0; u < 5; ++u) {
      float a = 0.0f;
#pragma unroll
      for (int j = 0; j < 8; ++j) a += q[s][j] * k[u][j];
      sc[u] = a * INV_SQRT_D_C;
      mx = fmaxf(mx, sc[u]);
    }
    float w[5], ssum = 0.0f;
#pragma unroll
    for (int u = 0; u < 5; ++u) { w[u] = expf(sc[u] - mx); ssum += w[u]; }
    float inv = 1.0f / ssum;
    float o[8] = {};
#pragma unroll
    for (int u = 0; u < 5; ++u) {
      float ww = w[u] * inv;
#pragma unroll
      for (int j = 0; j < 8; ++j) o[j] += ww * v[u][j];
    }
    float* ob = obuf + ((long long)s * NI_C + n) * 64 + h * 8;
#pragma unroll
    for (int j = 0; j < 8; ++j) ob[j] = o[j];
  }
}

// ---------------- sum the S projected outputs ----------------
__global__ void sum_layers(const float* __restrict__ proj, float* __restrict__ osum) {
  long long i = (long long)blockIdx.x * blockDim.x + threadIdx.x;
  if (i >= NI_C * 64) return;
  float a = 0.0f;
#pragma unroll
  for (int s = 0; s < 5; ++s) a += proj[(long long)s * NI_C * 64 + i];
  osum[i] = a;
}

// ---------------- MLP head 64->32->16->8->4->2->1 (gelu between) ----------------
// First-layer weights (8 KB) staged into LDS with the Tensor Data Mover (TDM):
// D# per ISA cdna5 §8.3/8.4 — count=1, type=2(image), data_size=4B,
// tensor_dim0 = tile_dim0 = 2048, tensor_dim1 = tile_dim1 = 1, stride0 = 2048.
// clang-23 toolchain: 6-arg builtin (g0, g1, g2, g3, g4, cpol).
__global__ __launch_bounds__(128)
void mlp_head(const float* __restrict__ xin,
              const float* mw0, const float* mb0, const float* mw1, const float* mb1,
              const float* mw2, const float* mb2, const float* mw3, const float* mb3,
              const float* mw4, const float* mb4, const float* mw5, const float* mb5,
              float* __restrict__ out)
{
  __shared__ float w0[2048], w1[512], w2[128], w3[32], w4[8], w5[2];
  __shared__ float b0[32], b1[16], b2[8], b3[4], b4[2], b5[1];

  if (threadIdx.x < 32) {            // wave 0 issues one TDM op for w0
    unsigned lds_off = (unsigned)(unsigned long long)(uintptr_t)w0;
    unsigned long long ga = (unsigned long long)(uintptr_t)mw0;
    u32x4 g0;
    g0[0] = 1u;                                               // count=1, user descriptor
    g0[1] = lds_off;                                          // lds_addr
    g0[2] = (unsigned)(ga & 0xffffffffu);                     // global_addr[31:0]
    g0[3] = (unsigned)((ga >> 32) & 0x1ffffffu) | (2u << 30); // global_addr[56:32] | type=2
    i32x8 g1;
    g1[0] = (int)(2u << 16);            // data_size = 2 (4 bytes)
    g1[1] = (int)(2048u << 16);         // tensor_dim0[15:0] in bits 63:48
    g1[2] = (int)(1u << 16);            // tensor_dim0 hi=0 | tensor_dim1 lo16 = 1
    g1[3] = (int)(2048u << 16);         // tensor_dim1 hi=0 | tile_dim0 = 2048
    g1[4] = 1;                          // tile_dim1 = 1, tile_dim2 = 0
    g1[5] = 2048;                       // tensor_dim0_stride[31:0]
    g1[6] = 0;
    g1[7] = 0;
    i32x4 z4 = {0, 0, 0, 0};
    i32x8 z8 = {0, 0, 0, 0, 0, 0, 0, 0};
    __builtin_amdgcn_tensor_load_to_lds(g0, g1, z4, z4, z8, 0);
    __builtin_amdgcn_s_wait_tensorcnt(0);
  }
  for (int i = threadIdx.x; i < 512;  i += blockDim.x) w1[i] = mw1[i];
  for (int i = threadIdx.x; i < 128;  i += blockDim.x) w2[i] = mw2[i];
  if (threadIdx.x < 32) { w3[threadIdx.x] = mw3[threadIdx.x]; b0[threadIdx.x] = mb0[threadIdx.x]; }
  if (threadIdx.x < 16) b1[threadIdx.x] = mb1[threadIdx.x];
  if (threadIdx.x < 8)  { w4[threadIdx.x] = mw4[threadIdx.x]; b2[threadIdx.x] = mb2[threadIdx.x]; }
  if (threadIdx.x < 4)  b3[threadIdx.x] = mb3[threadIdx.x];
  if (threadIdx.x < 2)  { w5[threadIdx.x] = mw5[threadIdx.x]; b4[threadIdx.x] = mb4[threadIdx.x]; }
  if (threadIdx.x < 1)  b5[0] = mb5[0];
  __syncthreads();

  long long n = (long long)blockIdx.x * blockDim.x + threadIdx.x;
  if (n >= NI_C) return;
  float a[64];
  for (int c = 0; c < 64; ++c) a[c] = xin[n * 64 + c];
  float h0[32];
  for (int o = 0; o < 32; ++o) {
    float acc = b0[o];
    for (int c = 0; c < 64; ++c) acc += w0[o * 64 + c] * a[c];
    h0[o] = gelu_exact(acc);
  }
  float h1[16];
  for (int o = 0; o < 16; ++o) {
    float acc = b1[o];
    for (int c = 0; c < 32; ++c) acc += w1[o * 32 + c] * h0[c];
    h1[o] = gelu_exact(acc);
  }
  float h2[8];
  for (int o = 0; o < 8; ++o) {
    float acc = b2[o];
    for (int c = 0; c < 16; ++c) acc += w2[o * 16 + c] * h1[c];
    h2[o] = gelu_exact(acc);
  }
  float h3[4];
  for (int o = 0; o < 4; ++o) {
    float acc = b3[o];
    for (int c = 0; c < 8; ++c) acc += w3[o * 8 + c] * h2[c];
    h3[o] = gelu_exact(acc);
  }
  float h4[2];
  for (int o = 0; o < 2; ++o) {
    float acc = b4[o];
    for (int c = 0; c < 4; ++c) acc += w4[o * 4 + c] * h3[c];
    h4[o] = gelu_exact(acc);
  }
  out[n] = b5[0] + w5[0] * h4[0] + w5[1] * h4[1];
}

// =================================================================================
extern "C" void kernel_launch(void* const* d_in, const int* in_sizes, int n_in,
                              void* d_out, int out_size, void* d_ws, size_t ws_size,
                              hipStream_t stream)
{
  (void)in_sizes; (void)n_in; (void)out_size; (void)ws_size;
  // ---- inputs (setup_inputs dict order) ----
  const float* x_instr = (const float*)d_in[0];
  const float* x_var   = (const float*)d_in[1];
  const float* Wk_i = (const float*)d_in[2];
  const float* Wq_i = (const float*)d_in[3];
  const float* Wv_i = (const float*)d_in[4];
  const float* Wa_i = (const float*)d_in[5];
  const float* bk_i = (const float*)d_in[6];
  const float* bq_i = (const float*)d_in[7];
  const float* bv_i = (const float*)d_in[8];
  const float* ba_i = (const float*)d_in[9];
  const float* skip_i = (const float*)d_in[10];
  const float* Wk_v = (const float*)d_in[11];
  const float* Wq_v = (const float*)d_in[12];
  const float* Wv_v = (const float*)d_in[13];
  const float* Wa_v = (const float*)d_in[14];
  const float* bk_v = (const float*)d_in[15];
  const float* bq_v = (const float*)d_in[16];
  const float* bv_v = (const float*)d_in[17];
  const float* ba_v = (const float*)d_in[18];
  const float* skip_v = (const float*)d_in[19];
  const float* arel_vv = (const float*)d_in[20];
  const float* mrel_vv = (const float*)d_in[21];
  const float* prel_vv = (const float*)d_in[22];
  const float* arel_vi = (const float*)d_in[23];
  const float* mrel_vi = (const float*)d_in[24];
  const float* prel_vi = (const float*)d_in[25];
  const float* arel_iv = (const float*)d_in[26];
  const float* mrel_iv = (const float*)d_in[27];
  const float* prel_iv = (const float*)d_in[28];
  const float* Wqkv = (const float*)d_in[29];
  const float* bqkv = (const float*)d_in[30];
  const float* Wo   = (const float*)d_in[31];
  const float* bo   = (const float*)d_in[32];
  const float* mw0 = (const float*)d_in[33]; const float* mb0 = (const float*)d_in[34];
  const float* mw1 = (const float*)d_in[35]; const float* mb1 = (const float*)d_in[36];
  const float* mw2 = (const float*)d_in[37]; const float* mb2 = (const float*)d_in[38];
  const float* mw3 = (const float*)d_in[39]; const float* mb3 = (const float*)d_in[40];
  const float* mw4 = (const float*)d_in[41]; const float* mb4 = (const float*)d_in[42];
  const float* mw5 = (const float*)d_in[43]; const float* mb5 = (const float*)d_in[44];
  const long long* src_vv = (const long long*)d_in[45];
  const long long* dst_vv = (const long long*)d_in[46];
  const long long* src_vi = (const long long*)d_in[47];
  const long long* dst_vi = (const long long*)d_in[48];
  const long long* src_iv = (const long long*)d_in[49];
  const long long* dst_iv = (const long long*)d_in[50];

  // ---- workspace carve-out ----
  char* wbase = (char*)d_ws;
  size_t off = 0;
  auto WS = [&](size_t nfloats) -> float* {
    float* p = (float*)(wbase + off);
    off = (off + nfloats * sizeof(float) + 255) & ~(size_t)255;
    return p;
  };
  float* xi_a = WS(NI_C * 64); float* xi_b = WS(NI_C * 64);
  float* xv_a = WS(NV_C * 64); float* xv_b = WS(NV_C * 64);
  float* k_i = WS(NI_C * 64);  float* q_i = WS(NI_C * 64);  float* v_i = WS(NI_C * 64);
  float* k_v = WS(NV_C * 64);  float* q_v = WS(NV_C * 64);  float* v_v = WS(NV_C * 64);
  float* krel = WS(NV_C * 64); float* vrel = WS(NV_C * 64);
  float* agg_i = WS(NI_C * 64); float* agg_v = WS(NV_C * 64);
  unsigned* mkey = (unsigned*)WS(NV_C * 8);
  float* sbuf = WS(NV_C * 8);
  float* alpha = WS(EVV_C * 8);
  float* outs  = WS(5 * NI_C * 64);
  float* qkvjk = WS(5 * NI_C * 192);
  float* objk  = WS(5 * NI_C * 64);
  float* projjk= WS(5 * NI_C * 64);
  float* osum  = WS(NI_C * 64);

  // ---- launch helpers ----
  auto gemm = [&](const float* X, const float* W, const float* b, float* Y,
                  long long N, int Cout) {
    dim3 g((unsigned)(N / 64), (unsigned)(Cout / 64)), bl(128);
    gemm64_wmma<false, false><<<g, bl, 0, stream>>>(X, W, b, Y, nullptr, nullptr, nullptr, Cout);
  };
  auto gemm_gelu_skip = [&](const float* X, const float* W, const float* b, float* Y,
                            const float* Xold, const float* sk, float* Y2, long long N) {
    dim3 g((unsigned)(N / 64), 1), bl(128);
    gemm64_wmma<true, true><<<g, bl, 0, stream>>>(X, W, b, Y, Xold, sk, Y2, 64);
  };
  auto fill = [&](float* p, long long n, float v) {
    fill_f32<<<2048, 256, 0, stream>>>(p, n, v);
  };
  auto edge_pass = [&](const long long* src, const long long* dst, long long E,
                       const float* qd, const float* kr, const float* vr,
                       const float* prel, float* agg, long long Ndst) {
    int eb = (int)((E * 8 + 255) / 256);
    fill((float*)mkey, Ndst * 8, 0.0f);
    edge_alpha<<<eb, 256, 0, stream>>>(src, dst, qd, kr, prel, alpha, mkey, E);
    int nb = (int)((Ndst * 8 + 255) / 256);
    finalize_max<<<nb, 256, 0, stream>>>(mkey, sbuf, Ndst * 8);
    edge_expsum<<<eb, 256, 0, stream>>>(dst, alpha, (const float*)mkey, sbuf, E);
    edge_agg<<<eb, 256, 0, stream>>>(src, dst, alpha, sbuf, vr, agg, E);
  };

  const float* xi = x_instr;
  const float* xv = x_var;
  float* xi_next = xi_a; float* xv_next = xv_a;

  for (int l = 0; l < L_C; ++l) {
    const size_t wl = (size_t)l * 4096, blv = (size_t)l * 64;
    const size_t rl = (size_t)l * 512, pl = (size_t)l * 8;
    // K/Q/V projections (WMMA)
    gemm(xi, Wk_i + wl, bk_i + blv, k_i, NI_C, 64);
    gemm(xi, Wq_i + wl, bq_i + blv, q_i, NI_C, 64);
    gemm(xi, Wv_i + wl, bv_i + blv, v_i, NI_C, 64);
    gemm(xv, Wk_v + wl, bk_v + blv, k_v, NV_C, 64);
    gemm(xv, Wq_v + wl, bq_v + blv, q_v, NV_C, 64);
    gemm(xv, Wv_v + wl, bv_v + blv, v_v, NV_C, 64);

    // ---- edge type vv : var -> var ----
    rel_transform<<<(unsigned)(NV_C * 8 / 256), 256, 0, stream>>>(
        k_v, v_v, arel_vv + rl, mrel_vv + rl, krel, vrel, NV_C);
    fill(agg_v, NV_C * 64, 0.0f);
    edge_pass(src_vv, dst_vv, EVV_C, q_v, krel, vrel, prel_vv + pl, agg_v, NV_C);

    // ---- edge type iv : instr -> var (accumulates into agg_v, group='sum') ----
    rel_transform<<<(unsigned)(NI_C * 8 / 256), 256, 0, stream>>>(
        k_i, v_i, arel_iv + rl, mrel_iv + rl, krel, vrel, NI_C);
    edge_pass(src_iv, dst_iv, EIV_C, q_v, krel, vrel, prel_iv + pl, agg_v, NV_C);

    // ---- edge type vi : var -> instr ----
    rel_transform<<<(unsigned)(NV_C * 8 / 256), 256, 0, stream>>>(
        k_v, v_v, arel_vi + rl, mrel_vi + rl, krel, vrel, NV_C);
    fill(agg_i, NI_C * 64, 0.0f);
    edge_pass(src_vi, dst_vi, EVI_C, q_i, krel, vrel, prel_vi + pl, agg_i, NI_C);

    // ---- output projection: gelu(agg) @ Wa^T + ba, sigmoid-skip blend ----
    float* y2 = (l > 0) ? (outs + (size_t)(l - 1) * NI_C * 64) : nullptr;
    gemm_gelu_skip(agg_i, Wa_i + wl, ba_i + blv, xi_next, xi, skip_i + l, y2, NI_C);
    gemm_gelu_skip(agg_v, Wa_v + wl, ba_v + blv, xv_next, xv, skip_v + l, nullptr, NV_C);

    xi = xi_next; xv = xv_next;
    xi_next = (xi == xi_a) ? xi_b : xi_a;
    xv_next = (xv == xv_a) ? xv_b : xv_a;
  }

  // ---- jumping-knowledge MHA over the 5 stacked instr outputs ----
  gemm(outs, Wqkv, bqkv, qkvjk, 5 * NI_C, 192);
  jk_attention<<<(unsigned)(NI_C * 8 / 128), 128, 0, stream>>>(qkvjk, objk);
  gemm(objk, Wo, bo, projjk, 5 * NI_C, 64);
  sum_layers<<<(unsigned)(NI_C * 64 / 256), 256, 0, stream>>>(projjk, osum);

  // ---- MLP head ----
  mlp_head<<<(unsigned)(NI_C / 128), 128, 0, stream>>>(
      osum, mw0, mb0, mw1, mb1, mw2, mb2, mw3, mb3, mw4, mb4, mw5, mb5,
      (float*)d_out);
}
